// LSTMCell_41764261986629
// MI455X (gfx1250) — matrix operand order repaired and checked by reading, not compiled, not run on hardware
//
#include <hip/hip_runtime.h>

typedef __attribute__((ext_vector_type(16))) __bf16 v16bf;
typedef __attribute__((ext_vector_type(8)))  float  v8f;

#define BATCH   4096
#define HID     1024
#define K_PHASE 1024
#define BM      128
#define BN      32
#define BK      32
#define LDA_E   40      // padded LDS row stride in bf16 elements (80 B, 16B aligned)

__device__ __forceinline__ unsigned short f2bf(float f) {
    unsigned u = __float_as_uint(f);
    return (unsigned short)((u + 0x7FFFu + ((u >> 16) & 1u)) >> 16);  // RNE
}

// fp32 -> bf16 conversion, 8 elements/thread (memory-bound, negligible cost)
__global__ __launch_bounds__(256) void cvt_f32_to_bf16(const float* __restrict__ src,
                                                       unsigned short* __restrict__ dst,
                                                       int n8) {
    int i = blockIdx.x * 256 + threadIdx.x;
    if (i >= n8) return;
    const float4* s4 = (const float4*)src;
    float4 a = s4[2 * i + 0];
    float4 b = s4[2 * i + 1];
    uint4 o;
    o.x = (unsigned)f2bf(a.x) | ((unsigned)f2bf(a.y) << 16);
    o.y = (unsigned)f2bf(a.z) | ((unsigned)f2bf(a.w) << 16);
    o.z = (unsigned)f2bf(b.x) | ((unsigned)f2bf(b.y) << 16);
    o.w = (unsigned)f2bf(b.z) | ((unsigned)f2bf(b.w) << 16);
    ((uint4*)dst)[i] = o;
}

// Fused LSTM cell: ifgo GEMM (bf16 WMMA) + bias + activations + c/h update.
// Grid: (HID/BN, BATCH/BM) = (32, 32). Block: 256 threads = 8 waves.
// Wave w: gate = w&3, m-half = w>>2. Each wave: 64x32 tile of one gate.
// Global->LDS staging uses CDNA5 async loads (ASYNCcnt), no VGPR staging.
__global__ __launch_bounds__(256) void lstm_gemm_fused(
    const unsigned short* __restrict__ Hbf,
    const unsigned short* __restrict__ Xbf,
    const unsigned short* __restrict__ Whbf,
    const unsigned short* __restrict__ Wxbf,
    const float* __restrict__ bh,
    const float* __restrict__ Cin,
    float* __restrict__ Hout,
    float* __restrict__ Cout)
{
    __shared__ __align__(16) unsigned char smem[40960];

    const int tid  = threadIdx.x;
    const int lane = tid & 31;
    const int wave = tid >> 5;
    const int gate = wave & 3;
    const int mh   = wave >> 2;          // which 64-row half of the 128-row block tile
    const int l15  = lane & 15;
    const int hiA  = (lane & 16) ? 8  : 0;   // A-frag K sub-offset per lane half
    const int hiB  = (lane & 16) ? 8  : 0;   // B-frag K sub-offset per lane half

    const int n0 = blockIdx.x * BN;      // column within one gate [0,1024)
    const int m0 = blockIdx.y * BM;      // batch row

    v8f acc[4][2] = {};

    // Per-thread constant part of the staging addresses (chunks c = tid, tid+256).
    const int c0row = tid >> 2, c0kq = tid & 3;                 // A chunk 0
    const int c1row = (tid + 256) >> 2, c1kq = tid & 3;         // A chunk 1 (+64 rows)
    const int g0 = tid >> 7, n0w = (tid >> 2) & 31;             // W chunk 0
    const int g1 = (tid + 256) >> 7, n1w = ((tid + 256) >> 2) & 31;

    const unsigned aLds0 = (unsigned)((c0row * LDA_E + c0kq * 8) * 2);
    const unsigned aLds1 = (unsigned)((c1row * LDA_E + c1kq * 8) * 2);
    const unsigned wLds0 = (unsigned)(20480 + ((g0 * 32 + n0w) * LDA_E + c0kq * 8) * 2);
    const unsigned wLds1 = (unsigned)(20480 + ((g1 * 32 + n1w) * LDA_E + c1kq * 8) * 2);

    // Issue 4 async b128 global->LDS transfers for tile kt into buffer buf.
    auto issue_async = [&](int kt, int buf) {
        const int phase = (kt >= 32);                     // 0: h/Wh, 1: x/Wx
        const int ks = kt * BK - phase * K_PHASE;         // phase-local k base
        const unsigned short* Asrc = phase ? Xbf : Hbf;
        const unsigned short* Wsrc = phase ? Wxbf : Whbf;
        const unsigned bofs = (unsigned)(buf * 10240);

        unsigned long long aG0 = (unsigned long long)(const char*)
            &Asrc[(size_t)(m0 + c0row) * K_PHASE + ks + c0kq * 8];
        unsigned long long aG1 = (unsigned long long)(const char*)
            &Asrc[(size_t)(m0 + c1row) * K_PHASE + ks + c1kq * 8];
        unsigned long long wG0 = (unsigned long long)(const char*)
            &Wsrc[(size_t)((g0 << 10) + n0 + n0w) * K_PHASE + ks + c0kq * 8];
        unsigned long long wG1 = (unsigned long long)(const char*)
            &Wsrc[(size_t)((g1 << 10) + n0 + n1w) * K_PHASE + ks + c1kq * 8];

        asm volatile("global_load_async_to_lds_b128 %0, %1, off"
                     :: "v"(aLds0 + bofs), "v"(aG0) : "memory");
        asm volatile("global_load_async_to_lds_b128 %0, %1, off"
                     :: "v"(aLds1 + bofs), "v"(aG1) : "memory");
        asm volatile("global_load_async_to_lds_b128 %0, %1, off"
                     :: "v"(wLds0 + bofs), "v"(wG0) : "memory");
        asm volatile("global_load_async_to_lds_b128 %0, %1, off"
                     :: "v"(wLds1 + bofs), "v"(wG1) : "memory");
    };

    issue_async(0, 0);
    asm volatile("s_wait_asynccnt 0x0" ::: "memory");
    __syncthreads();

    const int KT = 64;                   // 2048 / BK
    #pragma clang loop unroll(disable)
    for (int kt = 0; kt < KT; ++kt) {
        const int buf = kt & 1;
        if (kt + 1 < KT) issue_async(kt + 1, buf ^ 1);

        const unsigned short* As = (const unsigned short*)(smem + buf * 10240);
        const unsigned short* Ws = (const unsigned short*)(smem + 20480 + buf * 10240)
                                   + gate * 32 * LDA_E;

        union Frag { uint4 q[2]; v16bf v; };
        Frag bfr[2], afr[4];
        #pragma unroll
        for (int nt = 0; nt < 2; ++nt) {
            int n = nt * 16 + l15;       // B-frag: lane holds col n, K chunks {hiB, hiB+16}
            bfr[nt].q[0] = *(const uint4*)&Ws[n * LDA_E + hiB];
            bfr[nt].q[1] = *(const uint4*)&Ws[n * LDA_E + hiB + 16];
        }
        #pragma unroll
        for (int mt = 0; mt < 4; ++mt) {
            int r = mh * 64 + mt * 16 + l15;  // A-frag: lane holds row m, K chunks {hiA, hiA+16}
            afr[mt].q[0] = *(const uint4*)&As[r * LDA_E + hiA];
            afr[mt].q[1] = *(const uint4*)&As[r * LDA_E + hiA + 16];
        }
        #pragma unroll
        for (int mt = 0; mt < 4; ++mt)
            #pragma unroll
            for (int nt = 0; nt < 2; ++nt)
                acc[mt][nt] = __builtin_amdgcn_wmma_f32_16x16x32_bf16(
                    false, afr[mt].v, false, bfr[nt].v,
                    (short)0, acc[mt][nt], false, false);

        if (kt + 1 < KT) asm volatile("s_wait_asynccnt 0x0" ::: "memory");
        __syncthreads();
    }

    // ---- fused epilogue: two 64-row passes through LDS (reuses staging space)
    float* eg = (float*)smem;            // [4 gates][64 rows][33] fp32 = 33792 B
    const int EG_LD = 33;
    for (int p = 0; p < 2; ++p) {
        if (mh == p) {
            #pragma unroll
            for (int mt = 0; mt < 4; ++mt)
                #pragma unroll
                for (int nt = 0; nt < 2; ++nt)
                    #pragma unroll
                    for (int r = 0; r < 8; ++r) {
                        int row = mt * 16 + ((lane & 16) ? 8 : 0) + r;  // D-matrix layout
                        int col = nt * 16 + l15;
                        eg[(gate * 64 + row) * EG_LD + col] = acc[mt][nt][r];
                    }
        }
        __syncthreads();
        #pragma unroll
        for (int j = 0; j < 8; ++j) {
            int idx = tid + j * 256;     // 64 rows x 32 cols
            int row = idx >> 5, col = idx & 31;
            int gcol = n0 + col;
            float iv = eg[(0 * 64 + row) * EG_LD + col] + bh[gcol];
            float fv = eg[(1 * 64 + row) * EG_LD + col] + bh[HID + gcol];
            float gv = eg[(2 * 64 + row) * EG_LD + col] + bh[2 * HID + gcol];
            float ov = eg[(3 * 64 + row) * EG_LD + col] + bh[3 * HID + gcol];
            int grow = m0 + p * 64 + row;
            float cp = Cin[(size_t)grow * HID + gcol];
            float is = 1.0f / (1.0f + __expf(-iv));
            float fs = 1.0f / (1.0f + __expf(-fv));
            float os = 1.0f / (1.0f + __expf(-ov));
            float e2g = __expf(2.0f * gv);
            float gt  = (e2g - 1.0f) / (e2g + 1.0f);
            float cn  = fs * cp + is * gt;
            float e2c = __expf(2.0f * cn);
            float tc  = (e2c - 1.0f) / (e2c + 1.0f);
            Hout[(size_t)grow * HID + gcol] = os * tc;
            Cout[(size_t)grow * HID + gcol] = cn;
        }
        __syncthreads();
    }
}

extern "C" void kernel_launch(void* const* d_in, const int* in_sizes, int n_in,
                              void* d_out, int out_size, void* d_ws, size_t ws_size,
                              hipStream_t stream) {
    const float* x  = (const float*)d_in[0];   // [4096,1024]
    const float* h  = (const float*)d_in[1];   // [4096,1024]
    const float* c  = (const float*)d_in[2];   // [4096,1024]
    const float* Wh = (const float*)d_in[3];   // [4096,1024]
    const float* bh = (const float*)d_in[4];   // [4096]
    const float* Wx = (const float*)d_in[5];   // [4096,1024]

    const size_t NE = (size_t)4096 * 1024;     // 4,194,304 elements each
    unsigned short* Hbf  = (unsigned short*)d_ws;
    unsigned short* Xbf  = Hbf  + NE;
    unsigned short* Whbf = Xbf  + NE;
    unsigned short* Wxbf = Whbf + NE;          // total ws use: 32 MB

    const int n8 = (int)(NE / 8);              // 524288
    dim3 cgrd(n8 / 256), cblk(256);
    cvt_f32_to_bf16<<<cgrd, cblk, 0, stream>>>(h,  Hbf,  n8);
    cvt_f32_to_bf16<<<cgrd, cblk, 0, stream>>>(x,  Xbf,  n8);
    cvt_f32_to_bf16<<<cgrd, cblk, 0, stream>>>(Wh, Whbf, n8);
    cvt_f32_to_bf16<<<cgrd, cblk, 0, stream>>>(Wx, Wxbf, n8);

    float* Hout = (float*)d_out;
    float* Cout = Hout + NE;
    dim3 grid(HID / BN, BATCH / BM);           // (32, 32)
    lstm_gemm_fused<<<grid, dim3(256), 0, stream>>>(Hbf, Xbf, Whbf, Wxbf, bh, c, Hout, Cout);
}
